// CustomLayerModel_15625091023069
// MI455X (gfx1250) — compile-verified, matrix operand order checked by reading, not run on hardware
//
#include <hip/hip_runtime.h>
#include <hip/hip_bf16.h>

// ---------------------------------------------------------------------------
// CDNA5 / gfx1250 implementation of the 3-layer graph conv + GRU model.
// Dense GEMMs: v_wmma_f32_16x16x32_f16 (f16 in, f32 accumulate).
// GEMM tile staging: TDM tensor_load_to_lds (async tensor DMA, TENSORcnt).
// ---------------------------------------------------------------------------

typedef __attribute__((ext_vector_type(16))) _Float16 v16h;
typedef __attribute__((ext_vector_type(8)))  float    v8f;
typedef __attribute__((ext_vector_type(4)))  unsigned v4u;
typedef __attribute__((ext_vector_type(8)))  unsigned v8u;

#define HDIM 128
#define LDA  136   // LDS row pitch in halves (128 + 8 pad) == TDM pad 4 DW / 64 DW

// ---------------- elementwise helpers ----------------

__global__ void k_zero_f32(float* __restrict__ p, long n) {
    long i = (long)blockIdx.x * blockDim.x + threadIdx.x;
    if (i < n) p[i] = 0.0f;
}

__global__ void k_f32_to_f16(const float* __restrict__ s, _Float16* __restrict__ d, int n) {
    int i = blockIdx.x * blockDim.x + threadIdx.x;
    if (i < n) d[i] = (_Float16)s[i];
}

// degree accumulation: one thread per directed edge
__global__ void k_degree(const int* __restrict__ dst, float* __restrict__ deg, int nE) {
    int e = blockIdx.x * blockDim.x + threadIdx.x;
    if (e < nE) atomicAdd(&deg[dst[e]], 1.0f);
}

__global__ void k_invdeg(float* __restrict__ deg, int n) {
    int i = blockIdx.x * blockDim.x + threadIdx.x;
    if (i < n) deg[i] = 1.0f / fmaxf(deg[i], 1.0f);
}

// one wave per edge: 32 lanes x float4 = 128 channels
__global__ __launch_bounds__(256)
void k_edge_agg(const int* __restrict__ src, const int* __restrict__ dst,
                const float* __restrict__ x, float* __restrict__ agg, int nE) {
    int e = blockIdx.x * 8 + (threadIdx.x >> 5);
    if (e >= nE) return;
    const int s = src[e];
    const int d = dst[e];
    const int ch = (threadIdx.x & 31) * 4;
    const float4 v = *(const float4*)(x + (size_t)s * HDIM + ch);
    float* o = agg + (size_t)d * HDIM + ch;
    atomicAdd(o + 0, v.x);
    atomicAdd(o + 1, v.y);
    atomicAdd(o + 2, v.z);
    atomicAdd(o + 3, v.w);
}

// c_in = (x + agg * inv_deg) -> f16, one thread per float4 chunk
__global__ void k_fuse_input(const float* __restrict__ x, const float* __restrict__ agg,
                             const float* __restrict__ invdeg, _Float16* __restrict__ out,
                             int nNodes) {
    int i = blockIdx.x * blockDim.x + threadIdx.x;   // chunk id
    int total = nNodes * (HDIM / 4);
    if (i >= total) return;
    int row = i >> 5;               // HDIM/4 = 32 chunks per row
    int c4  = (i & 31) * 4;
    float inv = invdeg[row];
    const float4 xv = *(const float4*)(x   + (size_t)row * HDIM + c4);
    const float4 av = *(const float4*)(agg + (size_t)row * HDIM + c4);
    _Float16* o = out + (size_t)row * HDIM + c4;
    o[0] = (_Float16)(xv.x + av.x * inv);
    o[1] = (_Float16)(xv.y + av.y * inv);
    o[2] = (_Float16)(xv.z + av.z * inv);
    o[3] = (_Float16)(xv.w + av.w * inv);
}

// GRU gate math, one thread per (node, channel)
__global__ void k_gru(const _Float16* __restrict__ gi, const _Float16* __restrict__ gh,
                      float* __restrict__ h, _Float16* __restrict__ hf16, int nNodes) {
    long t = (long)blockIdx.x * blockDim.x + threadIdx.x;
    if (t >= (long)nNodes * HDIM) return;
    int n = (int)(t >> 7);
    int j = (int)(t & 127);
    const size_t gb = (size_t)n * (3 * HDIM) + j;
    float gir = (float)gi[gb];
    float giz = (float)gi[gb + HDIM];
    float gin = (float)gi[gb + 2 * HDIM];
    float ghr = (float)gh[gb];
    float ghz = (float)gh[gb + HDIM];
    float ghn = (float)gh[gb + 2 * HDIM];
    float r   = 1.0f / (1.0f + __expf(-(gir + ghr)));
    float z   = 1.0f / (1.0f + __expf(-(giz + ghz)));
    float nst = tanhf(gin + r * ghn);
    const size_t hb = (size_t)n * HDIM + j;
    float hprev = h[hb];
    float hn = (1.0f - z) * nst + z * hprev;
    h[hb]    = hn;
    hf16[hb] = (_Float16)hn;
}

// ---------------- TDM: DMA one 2D f16 tile (rows x 128) into LDS ----------------
// Loads a 128-row x 128-half tile starting at `gsrc`, row stride 128 halves,
// into LDS at byte offset `lds_addr` with a 16B pad every 256B (-> 136-half pitch).
// Rows >= rows_remaining are zero-filled by TDM OOB handling.
__device__ __forceinline__ void tdm_load_tile_f16(unsigned lds_addr, const _Float16* gsrc,
                                                  int rows_remaining) {
    unsigned long long ga = (unsigned long long)(const void*)gsrc;
    unsigned rr = (unsigned)(rows_remaining < 0 ? 0 : rows_remaining);
    v4u g0;
    g0[0] = 1u;                                            // count=1, user descriptor
    g0[1] = lds_addr;                                      // LDS byte address
    g0[2] = (unsigned)ga;                                  // global_addr[31:0]
    g0[3] = (unsigned)((ga >> 32) & 0x1FFFFFFu)            // global_addr[56:32]
            | (2u << 30);                                  // type = 2 ("image")
    v8u g1;
    g1[0] = (1u << 16)                                     // data_size = 2 bytes
          | (1u << 20)                                     // pad_enable
          | (5u << 22)                                     // pad_interval: 64 DWORDs (256B)
          | (3u << 25);                                    // pad_amount: 4 DWORDs (16B)
    g1[1] = (128u << 16);                                  // tensor_dim0 = 128 (bits 79:64? no: 63:48)
    g1[2] = (rr & 0xFFFFu) << 16;                          // tensor_dim1[15:0] @ bits 95:80
    g1[3] = (128u << 16) | ((rr >> 16) & 0xFFFFu);         // tile_dim0=128, tensor_dim1[31:16]
    g1[4] = 128u;                                          // tile_dim1 = 128, tile_dim2 = 0
    g1[5] = 128u;                                          // tensor_dim0_stride = 128
    g1[6] = 0u;
    g1[7] = 0u;
    asm volatile("tensor_load_to_lds %0, %1" :: "s"(g0), "s"(g1) : "memory");
}

// ---------------- WMMA GEMM ----------------
// C[M, Nout] = A[M,128] @ Bw[Nout,128]^T  (+ bias, + epilogue)
// EPI: 0 = bias -> f16 ; 1 = bias+ReLU -> f16 ; 2 = bias+BN -> f16 ; 3 = bias -> f32
template <int EPI>
__global__ __launch_bounds__(256)
void k_gemm_wmma(const _Float16* __restrict__ A, const _Float16* __restrict__ Bw,
                 const float* __restrict__ bias, int M, int Nout, int ldc,
                 _Float16* __restrict__ Ch, float* __restrict__ Cf,
                 const float* __restrict__ bnG, const float* __restrict__ bnB,
                 const float* __restrict__ bnM, const float* __restrict__ bnV) {
    extern __shared__ __align__(16) char smem_raw[];
    _Float16* lA = (_Float16*)smem_raw;
    _Float16* lB = lA + 128 * LDA;

    const int tid  = threadIdx.x;
    const int wave = tid >> 5;
    const int lane = tid & 31;
    const int rowBase = blockIdx.y * 128;
    const int colBase = blockIdx.x * 128;

    // Wave 0 launches both tile DMAs on the Tensor Data Mover, then waits
    // TENSORcnt; the workgroup barrier publishes the LDS tiles to all waves.
    if (wave == 0) {
        tdm_load_tile_f16((unsigned)(size_t)lA, A + (size_t)rowBase * HDIM, M - rowBase);
        tdm_load_tile_f16((unsigned)(size_t)lB, Bw + (size_t)colBase * HDIM, Nout - colBase);
        __builtin_amdgcn_s_wait_tensorcnt(0);
    }
    __syncthreads();

    const int lm = lane & 15;
    const int g  = lane >> 4;

    v8f acc[8];
#pragma unroll
    for (int i = 0; i < 8; ++i) acc[i] = (v8f){0.f, 0.f, 0.f, 0.f, 0.f, 0.f, 0.f, 0.f};

    // K = 128 -> four 16x16x32 steps; A fragment reused across 8 column subtiles.
#pragma unroll
    for (int ks = 0; ks < 4; ++ks) {
        const int kBase = ks * 32;
        union { v16h v; unsigned u[8]; } fa;
        const unsigned* ap = (const unsigned*)(&lA[(wave * 16 + lm) * LDA + kBase]);
#pragma unroll
        for (int v = 0; v < 4; ++v) {    // ISA 16-bit A layout: lanes 0-15 K=0..7,16..23 ; 16-31 +8
            fa.u[v]     = ap[4 * g + v];
            fa.u[v + 4] = ap[8 + 4 * g + v];
        }
#pragma unroll
        for (int n16 = 0; n16 < 8; ++n16) {
            union { v16h v; unsigned u[8]; } fb;
            const unsigned* bp = (const unsigned*)(&lB[(n16 * 16 + lm) * LDA + kBase]);
#pragma unroll
            for (int v = 0; v < 4; ++v) {
                fb.u[v]     = bp[4 * g + v];
                fb.u[v + 4] = bp[8 + 4 * g + v];
            }
            acc[n16] = __builtin_amdgcn_wmma_f32_16x16x32_f16(
                false, fa.v, false, fb.v, (short)0, acc[n16], false, false);
        }
    }

    // Epilogue. C/D layout: row-in-tile = v + 8*g, col-in-tile = lane&15.
#pragma unroll
    for (int n16 = 0; n16 < 8; ++n16) {
        int col = colBase + n16 * 16 + lm;
        if (col >= Nout) continue;
        float bs = bias[col];
        float ga = 0.f, bb = 0.f, mu = 0.f, iv = 0.f;
        if constexpr (EPI == 2) {
            ga = bnG[col]; bb = bnB[col]; mu = bnM[col];
            iv = rsqrtf(bnV[col] + 1e-5f);
        }
#pragma unroll
        for (int v = 0; v < 8; ++v) {
            int row = rowBase + wave * 16 + g * 8 + v;
            if (row >= M) continue;
            float val = acc[n16][v] + bs;
            if constexpr (EPI == 1) val = fmaxf(val, 0.0f);
            if constexpr (EPI == 2) val = ga * (val - mu) * iv + bb;
            if constexpr (EPI == 3) Cf[(size_t)row * ldc + col] = val;
            else                    Ch[(size_t)row * ldc + col] = (_Float16)val;
        }
    }
}

// ---------------------------------------------------------------------------

extern "C" void kernel_launch(void* const* d_in, const int* in_sizes, int n_in,
                              void* d_out, int out_size, void* d_ws, size_t ws_size,
                              hipStream_t stream) {
    const float* x0   = (const float*)d_in[0];
    const int*   srcI = (const int*)d_in[1];
    const int*   dstI = (const int*)d_in[2];
    const float* W1   = (const float*)d_in[3];
    const float* b1   = (const float*)d_in[4];
    const float* W2   = (const float*)d_in[5];
    const float* b2   = (const float*)d_in[6];
    const float* bnG  = (const float*)d_in[7];
    const float* bnB  = (const float*)d_in[8];
    const float* bnM  = (const float*)d_in[9];
    const float* bnV  = (const float*)d_in[10];
    const float* Wih  = (const float*)d_in[11];
    const float* Whh  = (const float*)d_in[12];
    const float* bih  = (const float*)d_in[13];
    const float* bhh  = (const float*)d_in[14];
    const float* LW1  = (const float*)d_in[15];
    const float* Lb1  = (const float*)d_in[16];
    const float* LW2  = (const float*)d_in[17];
    const float* Lb2  = (const float*)d_in[18];
    float* out = (float*)d_out;

    const int Nn   = in_sizes[0] / HDIM;   // 100000 nodes
    const int twoE = in_sizes[1];          // 3.2M directed edges
    const int OUTC = 64;
    const int L    = 3;

    // ---- workspace partition ----
    char* p = (char*)d_ws;
    auto alloc = [&](size_t bytes) -> char* {
        char* r = p;
        p += (bytes + 255) & ~(size_t)255;
        return r;
    };
    float*    invdeg = (float*)alloc((size_t)Nn * 4);
    float*    hbuf   = (float*)alloc((size_t)Nn * HDIM * 4);
    _Float16* gi     = (_Float16*)alloc((size_t)Nn * 3 * HDIM * 2);
    _Float16* gh     = (_Float16*)alloc((size_t)Nn * 3 * HDIM * 2);
    _Float16* aIn    = (_Float16*)alloc((size_t)Nn * HDIM * 2);
    _Float16* t1     = (_Float16*)alloc((size_t)Nn * HDIM * 2);
    _Float16* cbn    = (_Float16*)alloc((size_t)Nn * HDIM * 2);
    _Float16* hf16   = (_Float16*)alloc((size_t)Nn * HDIM * 2);
    _Float16* wW1    = (_Float16*)alloc((size_t)L * HDIM * HDIM * 2);
    _Float16* wW2    = (_Float16*)alloc((size_t)L * HDIM * HDIM * 2);
    _Float16* wWih   = (_Float16*)alloc((size_t)3 * HDIM * HDIM * 2);
    _Float16* wWhh   = (_Float16*)alloc((size_t)3 * HDIM * HDIM * 2);
    _Float16* wLW1   = (_Float16*)alloc((size_t)HDIM * HDIM * 2);
    _Float16* wLW2   = (_Float16*)alloc((size_t)OUTC * HDIM * 2);
    float*    agg    = (float*)gi;   // alias: agg dead before gi is produced

    const dim3 blk(256);
    const size_t shmem = (size_t)2 * 128 * LDA * sizeof(_Float16);
    auto cdiv = [](long a, long b) { return (unsigned)((a + b - 1) / b); };

    // ---- one-time prep ----
    k_f32_to_f16<<<cdiv(L * HDIM * HDIM, 256), blk, 0, stream>>>(W1, wW1, L * HDIM * HDIM);
    k_f32_to_f16<<<cdiv(L * HDIM * HDIM, 256), blk, 0, stream>>>(W2, wW2, L * HDIM * HDIM);
    k_f32_to_f16<<<cdiv(3 * HDIM * HDIM, 256), blk, 0, stream>>>(Wih, wWih, 3 * HDIM * HDIM);
    k_f32_to_f16<<<cdiv(3 * HDIM * HDIM, 256), blk, 0, stream>>>(Whh, wWhh, 3 * HDIM * HDIM);
    k_f32_to_f16<<<cdiv(HDIM * HDIM, 256), blk, 0, stream>>>(LW1, wLW1, HDIM * HDIM);
    k_f32_to_f16<<<cdiv(OUTC * HDIM, 256), blk, 0, stream>>>(LW2, wLW2, OUTC * HDIM);

    k_zero_f32<<<cdiv(Nn, 256), blk, 0, stream>>>(invdeg, Nn);
    k_degree<<<cdiv(twoE, 256), blk, 0, stream>>>(dstI, invdeg, twoE);
    k_invdeg<<<cdiv(Nn, 256), blk, 0, stream>>>(invdeg, Nn);

    k_zero_f32<<<cdiv((long)Nn * HDIM, 256), blk, 0, stream>>>(hbuf, (long)Nn * HDIM);
    k_zero_f32<<<cdiv((long)Nn * HDIM / 2, 256), blk, 0, stream>>>((float*)hf16, (long)Nn * HDIM / 2);

    const float* cur_x = x0;
    const dim3 g128(1, cdiv(Nn, 128));
    const dim3 g384(3, cdiv(Nn, 128));

    for (int l = 0; l < L; ++l) {
        // mean-aggregation over neighbors
        k_zero_f32<<<cdiv((long)Nn * HDIM, 256), blk, 0, stream>>>(agg, (long)Nn * HDIM);
        k_edge_agg<<<cdiv(twoE, 8), blk, 0, stream>>>(srcI, dstI, cur_x, agg, twoE);
        k_fuse_input<<<cdiv((long)Nn * 32, 256), blk, 0, stream>>>(cur_x, agg, invdeg, aIn, Nn);

        // MLP: t1 = ReLU(c @ W1^T + b1) ; cbn = BN(t1 @ W2^T + b2)
        k_gemm_wmma<1><<<g128, blk, shmem, stream>>>(aIn, wW1 + (size_t)l * HDIM * HDIM,
            b1 + l * HDIM, Nn, HDIM, HDIM, t1, nullptr, nullptr, nullptr, nullptr, nullptr);
        k_gemm_wmma<2><<<g128, blk, shmem, stream>>>(t1, wW2 + (size_t)l * HDIM * HDIM,
            b2 + l * HDIM, Nn, HDIM, HDIM, cbn, nullptr,
            bnG + l * HDIM, bnB + l * HDIM, bnM + l * HDIM, bnV + l * HDIM);

        // GRU matmuls: gi = cbn @ Wih^T + bih ; gh = h @ Whh^T + bhh
        k_gemm_wmma<0><<<g384, blk, shmem, stream>>>(cbn, wWih, bih, Nn, 3 * HDIM, 3 * HDIM,
            gi, nullptr, nullptr, nullptr, nullptr, nullptr);
        k_gemm_wmma<0><<<g384, blk, shmem, stream>>>(hf16, wWhh, bhh, Nn, 3 * HDIM, 3 * HDIM,
            gh, nullptr, nullptr, nullptr, nullptr, nullptr);

        // gates + hidden-state update; x <- h
        k_gru<<<cdiv((long)Nn * HDIM, 256), blk, 0, stream>>>(gi, gh, hbuf, hf16, Nn);
        cur_x = hbuf;
    }

    // final MLP: out = ReLU(h @ LW1^T + Lb1) @ LW2^T + Lb2
    k_gemm_wmma<1><<<g128, blk, shmem, stream>>>(hf16, wLW1, Lb1, Nn, HDIM, HDIM,
        t1, nullptr, nullptr, nullptr, nullptr, nullptr);
    k_gemm_wmma<3><<<dim3(1, cdiv(Nn, 128)), blk, shmem, stream>>>(t1, wLW2, Lb2, Nn, OUTC, OUTC,
        nullptr, out, nullptr, nullptr, nullptr, nullptr);
}
